// MultiHeadAttentionBlock_35888746725378
// MI455X (gfx1250) — compile-verified, matrix operand order checked by reading, not run on hardware
//
#include <hip/hip_runtime.h>

#define Bdim 4
#define Sdim 2048
#define Ddim 1024
#define Hdim 16

typedef __attribute__((ext_vector_type(16))) __bf16 v16bf;
typedef __attribute__((ext_vector_type(8)))  __bf16 v8bf;
typedef __attribute__((ext_vector_type(8)))  float  v8f;
typedef __attribute__((ext_vector_type(4)))  unsigned int u32x4;
typedef __attribute__((ext_vector_type(8)))  unsigned int u32x8;

// Build a 16x bf16 fragment from two 16-byte chunks (two b128 loads).
static __device__ __forceinline__ v16bf ld2(const __bf16* p0, const __bf16* p1) {
    v8bf lo = *(const v8bf*)p0;
    v8bf hi = *(const v8bf*)p1;
    return __builtin_shufflevector(lo, hi, 0,1,2,3,4,5,6,7,8,9,10,11,12,13,14,15);
}

static __device__ __forceinline__ v8f wmma_bf16(v16bf a, v16bf b, v8f c) {
    // D = A(16x32 bf16) * B(32x16 bf16) + C(16x16 f32)
    return __builtin_amdgcn_wmma_f32_16x16x32_bf16(false, a, false, b, (short)0, c, false, false);
}

// TDM: tensor_load_to_lds with D# groups 0 (4 SGPR) and 1 (8 SGPR); 2D max.
static __device__ __forceinline__ void tdm_load(u32x4 g0, u32x8 g1) {
    asm volatile("tensor_load_to_lds %0, %1" :: "s"(g0), "s"(g1) : "memory");
}

// ---------------------------------------------------------------------------
// C[M,N] = A[M,K] * W[N,K]^T  (torch Linear, bias-free). f32 in, OutT out.
// Block: 256 threads = 8 waves (2m x 4n), tile 64(m) x 128(n), K-step 32.
// ---------------------------------------------------------------------------
template <typename OutT>
__global__ __launch_bounds__(256) void gemm_xwT(const float* __restrict__ A,
                                                const float* __restrict__ W,
                                                OutT* __restrict__ C,
                                                int M, int N, int K) {
    __shared__ __bf16 lA[64][32];    // 4 KB
    __shared__ __bf16 lB[128][32];   // 8 KB
    const int tid  = threadIdx.x;
    const int wave = tid >> 5, lane = tid & 31;
    const int wm = wave >> 2, wn = wave & 3;
    const int bm = blockIdx.x * 64, bn = blockIdx.y * 128;
    const int ln = lane & 15;
    const int kh = (lane >> 4) << 3;       // per-lane K chunk offset (0 or 8)
    v8f acc[2][2] = {};
    for (int k0 = 0; k0 < K; k0 += 32) {
        const int ar = tid >> 2, ac = (tid & 3) << 3;
        const float* ap = A + (size_t)(bm + ar) * K + k0 + ac;
#pragma unroll
        for (int i = 0; i < 8; ++i) lA[ar][ac + i] = (__bf16)ap[i];
        const int br = tid >> 1, bc = (tid & 1) << 4;
        const float* wp = W + (size_t)(bn + br) * K + k0 + bc;
#pragma unroll
        for (int i = 0; i < 16; ++i) lB[br][bc + i] = (__bf16)wp[i];
        __syncthreads();
#pragma unroll
        for (int mi = 0; mi < 2; ++mi) {
            const __bf16* arow = &lA[wm * 32 + mi * 16 + ln][0];
            v16bf afr = ld2(arow + kh, arow + 16 + kh);
#pragma unroll
            for (int ni = 0; ni < 2; ++ni) {
                const __bf16* brow = &lB[wn * 32 + ni * 16 + ln][0];
                v16bf bfr = ld2(brow + kh, brow + 16 + kh);
                acc[mi][ni] = wmma_bf16(afr, bfr, acc[mi][ni]);
            }
        }
        __syncthreads();
    }
#pragma unroll
    for (int mi = 0; mi < 2; ++mi)
#pragma unroll
        for (int ni = 0; ni < 2; ++ni)
#pragma unroll
            for (int r = 0; r < 8; ++r) {
                int row = bm + wm * 32 + mi * 16 + r + ((lane >> 4) << 3);
                int col = bn + wn * 32 + ni * 16 + ln;
                C[(size_t)row * N + col] = (OutT)acc[mi][ni][r];
            }
}

// ---------------------------------------------------------------------------
// RoPE (theta=10000, interleaved rotate_half) + repack [B,S,H*64] -> [B,H,S,64]
// ---------------------------------------------------------------------------
__global__ __launch_bounds__(256) void rope_pack(const __bf16* __restrict__ X,
                                                 __bf16* __restrict__ out) {
    int idx = blockIdx.x * 256 + threadIdx.x;    // B*S*H*32 pairs
    int p = idx & 31;
    int h = (idx >> 5) & 15;
    int s = (idx >> 9) & 2047;
    int b = idx >> 20;
    const __bf16* src = X + (size_t)(b * Sdim + s) * Ddim + h * 64 + 2 * p;
    float x0 = (float)src[0], x1 = (float)src[1];
    float ang = (float)s * __powf(10000.f, -(float)(2 * p) * (1.f / 64.f));
    float c = __cosf(ang), sn = __sinf(ang);
    __bf16* dst = out + ((size_t)(b * Hdim + h) * Sdim + s) * 64 + 2 * p;
    dst[0] = (__bf16)(x0 * c - x1 * sn);
    dst[1] = (__bf16)(x1 * c + x0 * sn);
}

// Repack V: [B,S,H*64] -> [B,H,64,S] (key-dim contiguous for P*V B-fragments)
__global__ __launch_bounds__(256) void v_transpose_pack(const __bf16* __restrict__ X,
                                                        __bf16* __restrict__ out) {
    int idx = blockIdx.x * 256 + threadIdx.x;    // B*H*64*S
    int s = idx & 2047;
    int d = (idx >> 11) & 63;
    int h = (idx >> 17) & 15;
    int b = idx >> 21;
    out[idx] = X[(size_t)(b * Sdim + s) * Ddim + h * 64 + d];
}

// ---------------------------------------------------------------------------
// Flash attention. Block = 8 waves, one 16-row query tile per wave, all waves
// share one (b,h). Per 32-key step: wave0 issues two TDM tensor_load_to_lds
// (K tile 32x64 as 1-D 4KB copy; V^T tile 64x32 as 2-D tile, stride S),
// waits TENSORcnt, barrier; then QK^T (4 WMMA), online softmax (f32 VALU +
// shfl_xor row reductions), P C-layout->A-layout via wave-private LDS tile,
// P*V (4 WMMA). Output f32 [B,S,D].
// ---------------------------------------------------------------------------
__global__ __launch_bounds__(256) void flash_attn(const __bf16* __restrict__ Q,
                                                  const __bf16* __restrict__ Kc,
                                                  const __bf16* __restrict__ Vt,
                                                  const int* __restrict__ mask,
                                                  float* __restrict__ O) {
    __shared__ __align__(16) __bf16 ldsP[8][16][32];   // 8 KB wave-private
    __shared__ __align__(16) __bf16 ldsK[32][64];      // 4 KB shared K tile
    __shared__ __align__(16) __bf16 ldsV[64][32];      // 4 KB shared V^T tile
    const int wave = threadIdx.x >> 5, lane = threadIdx.x & 31;
    const int ln = lane & 15;
    const int kh = (lane >> 4) << 3;
    const int b = blockIdx.z, h = blockIdx.y;
    const int qtile = blockIdx.x * 8 + wave;
    const size_t bh = (size_t)(b * Hdim + h);
    const __bf16* qb = Q  + bh * Sdim * 64;
    const __bf16* kb = Kc + bh * Sdim * 64;
    const __bf16* vb = Vt + bh * 64 * Sdim;
    const int* mp = mask + b * Sdim;

    const __bf16* qr = qb + (size_t)(qtile * 16 + ln) * 64;
    v16bf a0 = ld2(qr + kh,      qr + 16 + kh);   // DK 0..31
    v16bf a1 = ld2(qr + 32 + kh, qr + 48 + kh);   // DK 32..63

    // Constant parts of the TDM D# descriptors (group 1).
    // K tile: 1-D copy of 32*64 bf16 (rows contiguous). data_size=2B.
    u32x8 g1K;
    g1K[0] = 1u << 16;                    // data_size=1 (2 bytes)
    g1K[1] = 0;                           // tensor_dim0[15:0]=0 (dim0 = 1<<20)
    g1K[2] = 16u | (1u << 16);            // tensor_dim0[31:16]=16, tensor_dim1=1
    g1K[3] = 2048u << 16;                 // tile_dim0 = 2048 elements
    g1K[4] = 1u;                          // tile_dim1 = 1, tile_dim2 = 0
    g1K[5] = 2048u;                       // tensor_dim0_stride (unused, 1 row)
    g1K[6] = 0; g1K[7] = 0;
    // V tile: 2-D, 64 rows (dims) x 32 cols (keys), row stride S.
    u32x8 g1V;
    g1V[0] = 1u << 16;                    // data_size=1 (2 bytes)
    g1V[1] = 2048u << 16;                 // tensor_dim0 = S = 2048
    g1V[2] = 64u << 16;                   // tensor_dim0[31:16]=0, tensor_dim1=64
    g1V[3] = 32u << 16;                   // tile_dim0 = 32
    g1V[4] = 64u;                         // tile_dim1 = 64, tile_dim2 = 0
    g1V[5] = 2048u;                       // tensor_dim0_stride = S
    g1V[6] = 0; g1V[7] = 0;
    const unsigned int ldsKoff = (unsigned int)(uintptr_t)&ldsK[0][0];
    const unsigned int ldsVoff = (unsigned int)(uintptr_t)&ldsV[0][0];

    float mx[8], lsum[8];
#pragma unroll
    for (int r = 0; r < 8; ++r) { mx[r] = -1e30f; lsum[r] = 0.f; }
    v8f acc[4] = {};
    const v8f vzero = {};

    for (int j = 0; j < Sdim; j += 32) {
        __syncthreads();                  // previous tile fully consumed
        if (wave == 0) {
            unsigned long long gaK = (unsigned long long)(uintptr_t)(kb + (size_t)j * 64);
            u32x4 g0K;
            g0K[0] = 1u;                  // count=1
            g0K[1] = ldsKoff;
            g0K[2] = (unsigned int)gaK;
            g0K[3] = (unsigned int)(gaK >> 32) | 0x80000000u;   // type=2
            tdm_load(g0K, g1K);
            unsigned long long gaV = (unsigned long long)(uintptr_t)(vb + j);
            u32x4 g0V;
            g0V[0] = 1u;
            g0V[1] = ldsVoff;
            g0V[2] = (unsigned int)gaV;
            g0V[3] = (unsigned int)(gaV >> 32) | 0x80000000u;
            tdm_load(g0V, g1V);
            __builtin_amdgcn_s_wait_tensorcnt(0);
        }
        __syncthreads();                  // K/V tile visible to all waves

        const __bf16* kr0 = &ldsK[ln][0];
        const __bf16* kr1 = &ldsK[16 + ln][0];
        v16bf k0lo = ld2(kr0 + kh,      kr0 + 16 + kh);
        v16bf k0hi = ld2(kr0 + 32 + kh, kr0 + 48 + kh);
        v16bf k1lo = ld2(kr1 + kh,      kr1 + 16 + kh);
        v16bf k1hi = ld2(kr1 + 32 + kh, kr1 + 48 + kh);
        v8f c0 = wmma_bf16(a0, k0lo, vzero);
        c0     = wmma_bf16(a1, k0hi, c0);
        v8f c1 = wmma_bf16(a0, k1lo, vzero);
        c1     = wmma_bf16(a1, k1hi, c1);

        const int m0 = mp[j + ln];
        const int m1 = mp[j + 16 + ln];
        float s0[8], s1[8];
#pragma unroll
        for (int r = 0; r < 8; ++r) {
            s0[r] = m0 ? c0[r] * 0.125f : -10000.f;   // 1/sqrt(64)
            s1[r] = m1 ? c1[r] * 0.125f : -10000.f;
        }
#pragma unroll
        for (int r = 0; r < 8; ++r) {
            float v = fmaxf(s0[r], s1[r]);            // row spans 16 lanes/half
            v = fmaxf(v, __shfl_xor(v, 1, 32));
            v = fmaxf(v, __shfl_xor(v, 2, 32));
            v = fmaxf(v, __shfl_xor(v, 4, 32));
            v = fmaxf(v, __shfl_xor(v, 8, 32));
            float mnew = fmaxf(mx[r], v);
            float corr = __expf(mx[r] - mnew);
            mx[r] = mnew;
            float p0 = __expf(s0[r] - mnew);
            float p1 = __expf(s1[r] - mnew);
            float rs = p0 + p1;
            rs += __shfl_xor(rs, 1, 32);
            rs += __shfl_xor(rs, 2, 32);
            rs += __shfl_xor(rs, 4, 32);
            rs += __shfl_xor(rs, 8, 32);
            lsum[r] = lsum[r] * corr + rs;
#pragma unroll
            for (int g = 0; g < 4; ++g) acc[g][r] *= corr;
            int prow = r + ((lane >> 4) << 3);
            ldsP[wave][prow][ln]      = (__bf16)p0;   // C-layout -> LDS
            ldsP[wave][prow][16 + ln] = (__bf16)p1;
        }
        // Reload P in A-fragment layout (DS ops same-wave are in-order).
        const __bf16* pr = &ldsP[wave][ln][0];
        v16bf pa = ld2(pr + kh, pr + 16 + kh);
#pragma unroll
        for (int g = 0; g < 4; ++g) {
            const __bf16* vr = &ldsV[g * 16 + ln][0];
            v16bf vfr = ld2(vr + kh, vr + 16 + kh);
            acc[g] = wmma_bf16(pa, vfr, acc[g]);
        }
    }
#pragma unroll
    for (int g = 0; g < 4; ++g)
#pragma unroll
        for (int r = 0; r < 8; ++r) {
            int row = qtile * 16 + r + ((lane >> 4) << 3);
            int col = h * 64 + g * 16 + ln;
            O[(size_t)(b * Sdim + row) * Ddim + col] = acc[g][r] / lsum[r];
        }
}

extern "C" void kernel_launch(void* const* d_in, const int* in_sizes, int n_in,
                              void* d_out, int out_size, void* d_ws, size_t ws_size,
                              hipStream_t stream) {
    const float* q   = (const float*)d_in[0];
    const float* k   = (const float*)d_in[1];
    const float* v   = (const float*)d_in[2];
    const int*   msk = (const int*)d_in[3];
    const float* w_q = (const float*)d_in[4];
    const float* w_k = (const float*)d_in[5];
    const float* w_v = (const float*)d_in[6];
    const float* w_o = (const float*)d_in[7];
    float* out = (float*)d_out;

    char* ws = (char*)d_ws;
    const size_t szH = (size_t)Bdim * Sdim * Ddim * sizeof(__bf16);   // 16 MB
    __bf16* Qp  = (__bf16*)(ws);                 // pre-RoPE Q (bf16)
    __bf16* Kp  = (__bf16*)(ws + 1 * szH);       // pre-RoPE K
    __bf16* Vp  = (__bf16*)(ws + 2 * szH);       // pre-pack V
    __bf16* qpk = (__bf16*)(ws + 3 * szH);       // [B,H,S,64] roped Q
    __bf16* kpk = (__bf16*)(ws + 4 * szH);       // [B,H,S,64] roped K
    __bf16* vtp = (__bf16*)(ws + 5 * szH);       // [B,H,64,S] V^T
    float*  attn = (float*)(ws);                 // 32 MB, reuses dead Qp+Kp

    const int M = Bdim * Sdim;                   // 8192
    dim3 gblk(256), ggrid(M / 64, Ddim / 128);
    gemm_xwT<__bf16><<<ggrid, gblk, 0, stream>>>(q, w_q, Qp, M, Ddim, Ddim);
    gemm_xwT<__bf16><<<ggrid, gblk, 0, stream>>>(k, w_k, Kp, M, Ddim, Ddim);
    gemm_xwT<__bf16><<<ggrid, gblk, 0, stream>>>(v, w_v, Vp, M, Ddim, Ddim);

    int ropeN = Bdim * Sdim * Hdim * 32;         // pairs
    rope_pack<<<ropeN / 256, 256, 0, stream>>>(Qp, qpk);
    rope_pack<<<ropeN / 256, 256, 0, stream>>>(Kp, kpk);
    int vpN = Bdim * Hdim * 64 * Sdim;
    v_transpose_pack<<<vpN / 256, 256, 0, stream>>>(Vp, vtp);

    dim3 fgrid(Sdim / 16 / 8, Hdim, Bdim);
    flash_attn<<<fgrid, 256, 0, stream>>>(qpk, kpk, vtp, msk, attn);

    gemm_xwT<float><<<ggrid, gblk, 0, stream>>>(attn, w_o, out, M, Ddim, Ddim);
    (void)in_sizes; (void)n_in; (void)out_size; (void)ws_size;
}
